// MemoryLayerAttention_65137474011993
// MI455X (gfx1250) — matrix operand order repaired and checked by reading, not compiled, not run on hardware
//
#include <hip/hip_runtime.h>
#include <math.h>

// MemoryLayerAttention for gfx1250 (MI455X, wave32, WMMA).
// One block per RNN sequence (N = 128); memory state lives in LDS for all 8 steps.
// All GEMMs use v_wmma_f32_16x16x32_f16 (f16 in, f32 accumulate).
// All weight B operands K-contiguous f16 in LDS -> ds_load_b128 pairs.
// Softmax row-sums computed by an all-ones-column WMMA (no shuffle reduction).

typedef __attribute__((ext_vector_type(16))) _Float16 v16h;
typedef __attribute__((ext_vector_type(8)))  float    v8f;

#define Bq   32
#define Qn   4
#define Dd   32
#define Vn   8
#define Nn   128
#define Rr   256
#define Cc   32
#define Ee   32
#define CHh  4
#define KDk  32
#define FFf  128
#define Ll   257        // 1 + R
#define LP   288        // padded to 18 tiles of 16
#define NT   18         // row tiles
#define NKC  9          // key chunks of 32
#define NW   8          // waves per block
#define TPB  256
#define SCL  0.17677669529663687f   // 1/sqrt(32)
#define PEC  0.28782313662425572f   // ln(10000)/32
#define PIH  1.5707963267948966f    // pi/2

__device__ __forceinline__ v8f vzero() {
  v8f z;
#pragma unroll
  for (int i = 0; i < 8; ++i) z[i] = 0.f;
  return z;
}

__device__ __forceinline__ v16h vones16() {
  v16h o;
#pragma unroll
  for (int e = 0; e < 16; ++e) o[e] = (_Float16)1.f;
  return o;
}

__device__ __forceinline__ v8f wmma16(v16h a, v16h b, v8f c) {
  return __builtin_amdgcn_wmma_f32_16x16x32_f16(false, a, false, b, (short)0, c, false, false);
}

// A operand: 16x32 f16, row-major in LDS. Contiguous k order used here (B loads match):
// lane = m + 16*g ; element e -> K = 16*g + e
__device__ __forceinline__ v16h load_Ah(const _Float16* src, int ld) {
  int lane = threadIdx.x & 31, m = lane & 15, g = lane >> 4;
  v16h a;
#pragma unroll
  for (int e = 0; e < 16; ++e) {
    int k = 16 * g + e;
    a[e] = src[m * ld + k];
  }
  return a;
}

__device__ __forceinline__ v16h load_Af(const float* src, int ld) {
  int lane = threadIdx.x & 31, m = lane & 15, g = lane >> 4;
  v16h a;
#pragma unroll
  for (int e = 0; e < 16; ++e) {
    int k = 16 * g + e;
    a[e] = (_Float16)src[m * ld + k];
  }
  return a;
}

// B operand from transposed (N-major, K-contiguous) f16 storage:
// B[k][n] = wT[(col0+n)*ldk + k0 + k]; lane n+16g reads K = k0+16g..k0+16g+15
// -> 16 contiguous f16 per lane (merges into ds_load_b128 pairs).
__device__ __forceinline__ v16h load_Bw(const _Float16* wT, int ldk, int col0, int k0) {
  int lane = threadIdx.x & 31, nn = lane & 15, g = lane >> 4;
  const _Float16* p = wT + (col0 + nn) * ldk + k0 + 16 * g;
  v16h b;
#pragma unroll
  for (int e = 0; e < 16; ++e) b[e] = p[e];
  return b;
}

// B operand = transpose of 16 key rows (row-major LxE): B[k][n] = src[n*ld + k]
// (also K-contiguous per lane)
__device__ __forceinline__ v16h load_Bt(const _Float16* src, int ld) {
  int lane = threadIdx.x & 31, nn = lane & 15, g = lane >> 4;
  const _Float16* p = src + nn * ld + 16 * g;
  v16h b;
#pragma unroll
  for (int e = 0; e < 16; ++e) b[e] = p[e];
  return b;
}

__device__ __forceinline__ float rmax16(float v) {
  v = fmaxf(v, __shfl_xor(v, 1, 16));
  v = fmaxf(v, __shfl_xor(v, 2, 16));
  v = fmaxf(v, __shfl_xor(v, 4, 16));
  v = fmaxf(v, __shfl_xor(v, 8, 16));
  return v;
}

// branch-free sinusoidal positional encoding: cos(x) = sin(x + pi/2)
__device__ __forceinline__ float pe_val(int pos, int c) {
  float freq = __expf(-PEC * (float)(c & ~1));
  return __sinf((float)pos * freq + (float)(c & 1) * PIH);
}

__device__ __forceinline__ float sigm(float x) { return 1.f / (1.f + __expf(-x)); }

__global__ void __launch_bounds__(TPB)
mla_kernel(const float* __restrict__ queries, const float* __restrict__ values,
           const float* __restrict__ W_in,  const float* __restrict__ b_in,
           const float* __restrict__ W_me,  const float* __restrict__ b_me,
           const float* __restrict__ Wq,    const float* __restrict__ bq,
           const float* __restrict__ Wk,    const float* __restrict__ bk,
           const float* __restrict__ Wv,    const float* __restrict__ bv,
           const float* __restrict__ Wo,    const float* __restrict__ bo,
           const float* __restrict__ W_f1,  const float* __restrict__ b_f1,
           const float* __restrict__ W_f2,  const float* __restrict__ b_f2,
           const float* __restrict__ ln_g,  const float* __restrict__ ln_b,
           const float* __restrict__ W_out, const float* __restrict__ b_out,
           const float* __restrict__ W_mi,  const float* __restrict__ b_mi,
           float* __restrict__ out)
{
  __shared__ float    s_mem[Rr * Cc];        // 32 KB  persistent memory state (f32)
  __shared__ _Float16 s_aug[LP * Ee];        // 18 KB  augmented tokens (f16)
  __shared__ _Float16 s_k[LP * Ee];          // 18 KB  keys / later: att (f16)
  __shared__ _Float16 s_vT[Ee * LP];         // 18 KB  V transposed / later: nf row-major
  __shared__ float    s_acc[LP * Ee];        // 36 KB  f32 accumulator / mi staging
  __shared__ _Float16 s_stg[NW * 16 * 32];   //  8 KB  per-wave f16 A-layout staging
  __shared__ _Float16 s_wme[32 * 32];        //  2 KB  W_me^T  [n(32)][k(32)]
  __shared__ _Float16 s_wq[CHh * KDk * 32];  //  8 KB  Wq^T    [n(128)][k(32)]
  __shared__ _Float16 s_wk[CHh * KDk * 32];  //  8 KB  Wk^T    [n(128)][k(32)]
  __shared__ _Float16 s_wv[CHh * KDk * 32];  //  8 KB  Wv^T    [n(128)][k(32)]
  __shared__ _Float16 s_wo[CHh * 32 * 32];   //  8 KB  Wo^T    [h][n(32)][k(32)]
  __shared__ _Float16 s_wf1[FFf * 32];       //  8 KB  W_f1^T  [n(128)][k(32)]
  __shared__ _Float16 s_wf2[32 * FFf];       //  8 KB  W_f2^T  [n(32)][k(128)]
  __shared__ _Float16 s_wmi[48 * 32];        //  3 KB  W_mi^T  [n(48,pad0)][k(32)]

  const int n    = blockIdx.x;          // sequence id 0..127
  const int bb_  = n / Qn;              // batch index
  const int tid  = threadIdx.x;
  const int wave = tid >> 5;
  const int lane = tid & 31;
  const int ln16 = lane & 15;
  const int g16  = lane >> 4;

  _Float16* sth  = s_stg + wave * 16 * 32;  // 16x32 f16 staging (A-layout source)
  _Float16* s_nf = s_vT;                    // nf row-major reuse after attention

  // ---- one-time init: memory state, aug padding, transposed f16 weights ----
  for (int i = tid; i < Rr * Cc; i += TPB) s_mem[i] = 1e-6f;
  for (int i = tid; i < (LP - Ll) * Ee; i += TPB) s_aug[Ll * Ee + i] = (_Float16)0.f;
  for (int i = tid; i < 32 * 32; i += TPB) {          // W_me (32x32) -> ^T
    int nn_ = i >> 5, k = i & 31;
    s_wme[i] = (_Float16)W_me[k * Ee + nn_];
  }
  for (int i = tid; i < CHh * KDk * 32; i += TPB) {   // Wq/Wk/Wv (32x128) -> ^T
    int nn_ = i >> 5, k = i & 31;
    s_wq[i] = (_Float16)Wq[k * (CHh * KDk) + nn_];
    s_wk[i] = (_Float16)Wk[k * (CHh * KDk) + nn_];
    s_wv[i] = (_Float16)Wv[k * (CHh * KDk) + nn_];
  }
  for (int i = tid; i < CHh * 32 * 32; i += TPB) {    // Wo (4,32,32) -> per-head ^T
    int h = i >> 10, rem = i & 1023, nn_ = rem >> 5, k = rem & 31;
    s_wo[i] = (_Float16)Wo[h * 1024 + k * 32 + nn_];
  }
  for (int i = tid; i < FFf * 32; i += TPB) {         // W_f1 (32x128) -> ^T
    int nn_ = i >> 5, k = i & 31;
    s_wf1[i] = (_Float16)W_f1[k * FFf + nn_];
  }
  for (int i = tid; i < 32 * FFf; i += TPB) {         // W_f2 (128x32) -> ^T
    int nn_ = i >> 7, k = i & 127;
    s_wf2[i] = (_Float16)W_f2[k * Ee + nn_];
  }
  for (int i = tid; i < 48 * 32; i += TPB) {          // W_mi (32x33) -> ^T, pad to 48
    int nn_ = i >> 5, k = i & 31;
    s_wmi[i] = (nn_ < 1 + Cc) ? (_Float16)W_mi[k * (1 + Cc) + nn_] : (_Float16)0.f;
  }
  __syncthreads();

  for (int step = 0; step < Vn; ++step) {
    // ---- phase 1: zero acc, build aug = [ei ; mem @ W_me + b_me] + pe ----
    for (int i = tid; i < LP * Ee; i += TPB) s_acc[i] = 0.f;
    if (tid < Ee) {
      float s = b_in[tid];
      for (int j = 0; j < Dd; ++j) s += queries[n * Dd + j] * W_in[j * Ee + tid];
      const float* vrow = values + (bb_ * Vn + step) * Dd;
      for (int j = 0; j < Dd; ++j) s += vrow[j] * W_in[(Dd + j) * Ee + tid];
      s_aug[tid] = (_Float16)(s + pe_val(0, tid));
    }
    for (int job = wave; job < 32; job += NW) {          // 16 tiles x 2 N-halves
      int t = job >> 1, nh = job & 1;
      v16h A = load_Af(s_mem + t * 16 * Cc, Cc);
      v8f  c = wmma16(A, load_Bw(s_wme, 32, nh * 16, 0), vzero());
#pragma unroll
      for (int r = 0; r < 8; ++r) {
        int row = 1 + t * 16 + r + 8 * g16;
        int col = nh * 16 + ln16;
        s_aug[row * Ee + col] = (_Float16)(c[r] + b_me[col] + pe_val(row, col));
      }
    }
    __syncthreads();

    // ---- phase 2: multi-head attention, heads sequential ----
    for (int h = 0; h < CHh; ++h) {
      // K (row-major) and V (transposed) projections: 18 tiles x {k,v} x 2 N-halves
      for (int job = wave; job < NT * 4; job += NW) {
        int t = job >> 2, sel = job & 3, mat = sel >> 1, nh = sel & 1;
        v16h A = load_Ah(s_aug + t * 16 * Ee, Ee);
        const _Float16* W = mat ? s_wv : s_wk;
        const float*   bi = mat ? bv : bk;
        v8f c = wmma16(A, load_Bw(W, 32, h * KDk + nh * 16, 0), vzero());
#pragma unroll
        for (int r = 0; r < 8; ++r) {
          int row = t * 16 + r + 8 * g16;
          int col = nh * 16 + ln16;
          float val = c[r] + bi[h * KDk + col];
          if (mat == 0) s_k[row * Ee + col] = (_Float16)val;   // keys row-major
          else          s_vT[col * LP + row] = (_Float16)val;  // V transposed (b128 store)
        }
      }
      __syncthreads();

      // flash attention per 16-row query tile; one wave per tile
      for (int qt = wave; qt < NT; qt += NW) {
        // q tile = aug_tile @ Wq_h + bq  -> stage -> A operand
        v16h Aau = load_Ah(s_aug + qt * 16 * Ee, Ee);
        v8f q0 = wmma16(Aau, load_Bw(s_wq, 32, h * KDk,      0), vzero());
        v8f q1 = wmma16(Aau, load_Bw(s_wq, 32, h * KDk + 16, 0), vzero());
#pragma unroll
        for (int r = 0; r < 8; ++r) {
          int rr = r + 8 * g16;
          sth[rr * 32 + ln16]      = (_Float16)(q0[r] + bq[h * KDk + ln16]);
          sth[rr * 32 + 16 + ln16] = (_Float16)(q1[r] + bq[h * KDk + 16 + ln16]);
        }
        v16h Aq = load_Ah(sth, 32);

        float mr[8];
        v8f o0 = vzero(), o1 = vzero(), o2 = vzero();   // o2: row-sum accumulator
#pragma unroll
        for (int r = 0; r < 8; ++r) mr[r] = -1e30f;

        for (int kc = 0; kc < NKC; ++kc) {
          v8f s0 = wmma16(Aq, load_Bt(s_k + (kc * 32)      * Ee, Ee), vzero());
          v8f s1 = wmma16(Aq, load_Bt(s_k + (kc * 32 + 16) * Ee, Ee), vzero());
          int k0 = kc * 32 + ln16, k1 = k0 + 16;
          float p0[8], p1[8];
#pragma unroll
          for (int r = 0; r < 8; ++r) {
            float a = (k0 < Ll) ? s0[r] * SCL : -1e30f;
            float b = (k1 < Ll) ? s1[r] * SCL : -1e30f;
            float nm = fmaxf(mr[r], rmax16(fmaxf(a, b)));
            float al = __expf(mr[r] - nm);
            mr[r] = nm;
            p0[r] = __expf(a - nm);
            p1[r] = __expf(b - nm);
            o0[r] *= al;
            o1[r] *= al;
            o2[r] *= al;
          }
#pragma unroll
          for (int r = 0; r < 8; ++r) {     // stage P (C-layout -> A-layout via LDS)
            int rr = r + 8 * g16;
            sth[rr * 32 + ln16]      = (_Float16)p0[r];
            sth[rr * 32 + 16 + ln16] = (_Float16)p1[r];
          }
          v16h Ap = load_Ah(sth, 32);
          o0 = wmma16(Ap, load_Bw(s_vT, LP, 0,  kc * 32), o0);
          o1 = wmma16(Ap, load_Bw(s_vT, LP, 16, kc * 32), o1);
          o2 = wmma16(Ap, vones16(), o2);   // every column = row-sum of P chunk
        }

        // normalize, apply Wo_h, accumulate into f32 attention output
#pragma unroll
        for (int r = 0; r < 8; ++r) {
          float inv = 1.f / fmaxf(o2[r], 1e-30f);   // softmax denominator, per lane
          int rr = r + 8 * g16;
          sth[rr * 32 + ln16]      = (_Float16)(o0[r] * inv);
          sth[rr * 32 + 16 + ln16] = (_Float16)(o1[r] * inv);
        }
        v16h Ao = load_Ah(sth, 32);
        v8f w0 = wmma16(Ao, load_Bw(s_wo + h * 1024, 32, 0,  0), vzero());
        v8f w1 = wmma16(Ao, load_Bw(s_wo + h * 1024, 32, 16, 0), vzero());
#pragma unroll
        for (int r = 0; r < 8; ++r) {
          int row = qt * 16 + r + 8 * g16;
          s_acc[row * Ee + ln16]      += w0[r];
          s_acc[row * Ee + 16 + ln16] += w1[r];
        }
      }
      __syncthreads();
    }

    // ---- phase 3: att = LN(mha + bo + aug); keep f32 in s_acc, f16 in s_k ----
    for (int row = tid; row < LP; row += TPB) {
      float su = 0.f, sq = 0.f;
      for (int c = 0; c < Ee; ++c) {
        float x = s_acc[row * Ee + c] + bo[c] + (float)s_aug[row * Ee + c];
        su += x; sq += x * x;
      }
      float mean = su * (1.f / 32.f);
      float var  = fmaxf(sq * (1.f / 32.f) - mean * mean, 0.f);
      float inv  = rsqrtf(var + 1e-6f);
      for (int c = 0; c < Ee; ++c) {
        float x = s_acc[row * Ee + c] + bo[c] + (float)s_aug[row * Ee + c];
        float a = (x - mean) * inv * ln_g[c] + ln_b[c];
        s_k[row * Ee + c]   = (_Float16)a;   // att f16 (WMMA A operand)
        s_acc[row * Ee + c] = a;             // att f32 (residual)
      }
    }
    __syncthreads();

    // ---- phase 4: FF  h = relu(att @ W_f1 + b_f1) @ W_f2 + b_f2 + att ----
    for (int t = wave; t < NT; t += NW) {
      v16h Aat = load_Ah(s_k + t * 16 * Ee, Ee);
      v8f h0 = vzero(), h1 = vzero();
      for (int kc = 0; kc < 4; ++kc) {             // FF hidden in 32-wide K chunks
        v8f f0 = wmma16(Aat, load_Bw(s_wf1, 32, kc * 32,      0), vzero());
        v8f f1 = wmma16(Aat, load_Bw(s_wf1, 32, kc * 32 + 16, 0), vzero());
#pragma unroll
        for (int r = 0; r < 8; ++r) {
          int rr = r + 8 * g16;
          sth[rr * 32 + ln16]      = (_Float16)fmaxf(f0[r] + b_f1[kc * 32 + ln16], 0.f);
          sth[rr * 32 + 16 + ln16] = (_Float16)fmaxf(f1[r] + b_f1[kc * 32 + 16 + ln16], 0.f);
        }
        v16h Ah = load_Ah(sth, 32);
        h0 = wmma16(Ah, load_Bw(s_wf2, FFf, 0,  kc * 32), h0);
        h1 = wmma16(Ah, load_Bw(s_wf2, FFf, 16, kc * 32), h1);
      }
#pragma unroll
      for (int r = 0; r < 8; ++r) {
        int row = t * 16 + r + 8 * g16;
        s_acc[row * Ee + ln16]      = h0[r] + b_f2[ln16]      + s_acc[row * Ee + ln16];
        s_acc[row * Ee + 16 + ln16] = h1[r] + b_f2[16 + ln16] + s_acc[row * Ee + 16 + ln16];
      }
    }
    __syncthreads();

    // ---- phase 5: nf = LN(h) -> f16 row-major in s_nf (reuses s_vT) ----
    for (int row = tid; row < LP; row += TPB) {
      const float* src = s_acc + row * Ee;
      float su = 0.f, sq = 0.f;
      for (int c = 0; c < Ee; ++c) { float x = src[c]; su += x; sq += x * x; }
      float mean = su * (1.f / 32.f);
      float var  = fmaxf(sq * (1.f / 32.f) - mean * mean, 0.f);
      float inv  = rsqrtf(var + 1e-6f);
      for (int c = 0; c < Ee; ++c)
        s_nf[row * Ee + c] = (_Float16)((src[c] - mean) * inv * ln_g[c] + ln_b[c]);
    }
    __syncthreads();

    // ---- phase 6: output scalar (last step only) + gated memory update ----
    // s_acc is dead here (nf extracted) -> reuse as per-wave 16x48 f32 mi staging
    if (step == Vn - 1 && wave == 0) {
      float p = (float)s_nf[lane] * W_out[lane];
#pragma unroll
      for (int m2 = 16; m2 >= 1; m2 >>= 1) p += __shfl_xor(p, m2, 32);
      if (lane == 0) out[n] = p + b_out[0];
    }
    {
      float* mif = s_acc + wave * (16 * 48);        // 768 f32 per wave, fits in s_acc
      for (int t = wave; t < 16; t += NW) {         // 16 tiles of memory rows
        v16h A  = load_Ah(s_nf + (1 + t * 16) * Ee, Ee);   // nf rows 1..256
        v8f m0 = wmma16(A, load_Bw(s_wmi, 32, 0,  0), vzero());
        v8f m1 = wmma16(A, load_Bw(s_wmi, 32, 16, 0), vzero());
        v8f m2 = wmma16(A, load_Bw(s_wmi, 32, 32, 0), vzero());
#pragma unroll
        for (int r = 0; r < 8; ++r) {               // stage mi tile (16x33 f32)
          int rr = r + 8 * g16;
          mif[rr * 48 + ln16]      = m0[r] + b_mi[ln16];
          mif[rr * 48 + 16 + ln16] = m1[r] + b_mi[16 + ln16];
          mif[rr * 48 + 32 + ln16] = (ln16 == 0) ? (m2[r] + b_mi[32]) : 0.f;
        }
        for (int idx = lane; idx < 16 * Cc; idx += 32) {
          int r = idx >> 5, cc = idx & 31;
          float cs   = sigm(mif[r * 48]);           // sigmoid(mi[...,0])
          float cand = mif[r * 48 + 1 + cc];
          int   mrow = t * 16 + r;
          float mold = s_mem[mrow * Cc + cc];
          s_mem[mrow * Cc + cc] = sigm(-cs) * mold + sigm(cs) * cand;
        }
      }
    }
    __syncthreads();
  }
}

extern "C" void kernel_launch(void* const* d_in, const int* in_sizes, int n_in,
                              void* d_out, int out_size, void* d_ws, size_t ws_size,
                              hipStream_t stream) {
  (void)in_sizes; (void)n_in; (void)d_ws; (void)ws_size; (void)out_size;
  const float* P[24];
  for (int i = 0; i < 24; ++i) P[i] = (const float*)d_in[i];
  mla_kernel<<<Nn, TPB, 0, stream>>>(
      P[0], P[1], P[2], P[3], P[4], P[5], P[6], P[7], P[8], P[9], P[10], P[11],
      P[12], P[13], P[14], P[15], P[16], P[17], P[18], P[19], P[20], P[21], P[22], P[23],
      (float*)d_out);
}
